// VolumeFeatureRenderer_20435454394806
// MI455X (gfx1250) — matrix operand order repaired and checked by reading, not compiled, not run on hardware
//
#include <hip/hip_runtime.h>
#include <hip/hip_bf16.h>
#include <stdint.h>

typedef __attribute__((ext_vector_type(16))) _Float16 v16h;
typedef __attribute__((ext_vector_type(8)))  float    v8f;
typedef __attribute__((ext_vector_type(4)))  unsigned int u32x4;
typedef __attribute__((ext_vector_type(8)))  int          i32x8;
typedef __attribute__((ext_vector_type(4)))  int          i32x4;

#define NBATCH 2
#define NHH 64
#define NWW 64
#define NSAMP 18
#define NRAYS 8192
#define HID 256
#define EMBC 32
#define SSZ 256

// output offsets (floats), reference return order:
// rgb_map(2,64,64,3), feature_map(2,64,64,256), sdf(2,64,64,18,1), mask(2,64,64,1), xyz(2,64,64,3)
#define OUT_RGB  0
#define OUT_FEAT 24576
#define OUT_SDF  2121728
#define OUT_MASK 2269184
#define OUT_XYZ  2277376

// packed f16 weight offsets (in halves) inside workspace Wp region
#define WP_L0 0        // Kt=1  (K 3  -> 32)
#define WP_L1 8192     // Kt=8  (K 256)
#define WP_L2 73728    // Kt=8  (K 256)
#define WP_LV 139264   // Kt=10 (K 291 -> 320)
#define WP_TOTAL 221184

__device__ __forceinline__ int iminD(int a, int b) { return a < b ? a : b; }
__device__ __forceinline__ int imaxD(int a, int b) { return a > b ? a : b; }

// ---------------------------------------------------------------------------
// Prep kernel 1: FiLM gamma/beta per (batch, layer, out-channel)
// gamma = 15*(style @ gW.T + gb) + 30 ; beta = 0.25*(style @ bW.T + bb)
// ---------------------------------------------------------------------------
__global__ void film_prep(const float* __restrict__ styles,
                          const float* gW0, const float* gb0, const float* bW0, const float* bb0,
                          const float* gW1, const float* gb1, const float* bW1, const float* bb1,
                          const float* gW2, const float* gb2, const float* bW2, const float* bb2,
                          const float* gW3, const float* gb3, const float* bW3, const float* bb3,
                          float* __restrict__ gam, float* __restrict__ bet) {
  int idx = blockIdx.x * blockDim.x + threadIdx.x;
  if (idx >= NBATCH * 4 * HID) return;
  int o = idx & 255;
  int L = (idx >> 8) & 3;
  int b = idx >> 10;
  const float *gW, *gb, *bW, *bb;
  if (L == 0)      { gW = gW0; gb = gb0; bW = bW0; bb = bb0; }
  else if (L == 1) { gW = gW1; gb = gb1; bW = bW1; bb = bb1; }
  else if (L == 2) { gW = gW2; gb = gb2; bW = bW2; bb = bb2; }
  else             { gW = gW3; gb = gb3; bW = bW3; bb = bb3; }
  const float* st = styles + (size_t)(b * 4 + L) * HID;
  float dg = 0.f, db = 0.f;
  for (int k = 0; k < HID; ++k) {
    float s = st[k];
    dg += s * gW[(size_t)o * HID + k];
    db += s * bW[(size_t)o * HID + k];
  }
  gam[idx] = 15.f * (dg + gb[o]) + 30.f;
  bet[idx] = 0.25f * (db + bb[o]);
}

// ---------------------------------------------------------------------------
// Prep kernel 2: pack weights into f16 WMMA B-fragment-major layout:
// Wp[kt][nt][lane][e] with e=0..15, K = kt*32 + (lane>=16?16:0) + e,
// n = nt*16 + (lane&15); value = W[n][K] (zero pad for K >= Kreal).
// ---------------------------------------------------------------------------
__global__ void pack_weights(const float* __restrict__ W0, const float* __restrict__ W1,
                             const float* __restrict__ W2, const float* __restrict__ WV,
                             _Float16* __restrict__ Wp) {
  int idx = blockIdx.x * blockDim.x + threadIdx.x;
  if (idx >= WP_TOTAL) return;
  int base, Kreal;
  const float* src;
  if (idx < WP_L1)      { base = WP_L0; Kreal = 3;   src = W0; }
  else if (idx < WP_L2) { base = WP_L1; Kreal = 256; src = W1; }
  else if (idx < WP_LV) { base = WP_L2; Kreal = 256; src = W2; }
  else                  { base = WP_LV; Kreal = 291; src = WV; }
  int rel = idx - base;
  int kt   = rel >> 13;        // /8192 halves per k-tile slab
  int rem  = rel & 8191;
  int nt   = rem >> 9;         // /512 halves per n-tile
  int rem2 = rem & 511;
  int ln   = rem2 >> 4;
  int e    = rem2 & 15;
  int K = kt * 32 + ((ln >= 16) ? 16 : 0) + e;
  int n = nt * 16 + (ln & 15);
  float v = (K < Kreal) ? src[(size_t)n * Kreal + K] : 0.f;
  Wp[idx] = (_Float16)v;
}

// ---------------------------------------------------------------------------
// Main kernel: 1 wave = 1 ray (18 samples -> two 16-row WMMA tiles).
// Block = 128 threads = 4 waves = 4 rays. Grid = 2048 blocks.
// Weight slabs (16KB per k-tile) are fetched global->LDS by the Tensor Data
// Mover (one TENSOR_LOAD_TO_LDS per k-tile, issued by wave 0, tracked with
// TENSORcnt), then consumed as WMMA B fragments by all 4 waves.
// ---------------------------------------------------------------------------
__launch_bounds__(128, 1)
__global__ void render_kernel(
    const float* __restrict__ pts, const float* __restrict__ rays_d,
    const float* __restrict__ viewdirs, const float* __restrict__ z_vals,
    const float* __restrict__ nearp, const float* __restrict__ farp,
    const float* __restrict__ planes, const float* __restrict__ plane_axes,
    const float* __restrict__ sbeta,
    const float* __restrict__ bias0, const float* __restrict__ bias1,
    const float* __restrict__ bias2, const float* __restrict__ biasv,
    const float* __restrict__ rgbW, const float* __restrict__ rgbB,
    const float* __restrict__ sigW, const float* __restrict__ sigB,
    const float* __restrict__ gam, const float* __restrict__ bet,
    const _Float16* __restrict__ Wp,
    float* __restrict__ out) {
  // LDS: 16KB weight slab + 4 * (16x320 f16) activation tiles + scan buffers
  __shared__ __align__(16) _Float16 sW[8192];
  __shared__ __align__(16) _Float16 sAct[4][16 * 320];
  __shared__ float sScan[4][16];
  __shared__ float sWgt[4][16];

  const int wave = threadIdx.x >> 5;
  const int lane = threadIdx.x & 31;
  const int ray  = blockIdx.x * 4 + wave;
  const int bidx = ray >> 12;              // ray / (64*64)
  _Float16* act = sAct[wave];

  const float nearv = nearp[0], farv = farp[0];
  const float inv_fn = 2.f / (farv - nearv);
  const float beta = sbeta[0];

  const float rdx = rays_d[ray * 3 + 0], rdy = rays_d[ray * 3 + 1], rdz = rays_d[ray * 3 + 2];
  const float nrd = __builtin_sqrtf(rdx * rdx + rdy * rdy + rdz * rdz);
  const float vdx = viewdirs[ray * 3 + 0], vdy = viewdirs[ray * 3 + 1], vdz = viewdirs[ray * 3 + 2];

  // invert the 3 plane-axes matrices (adjugate); only columns 0,1 are used
  float inv[3][3][2];
#pragma unroll
  for (int p = 0; p < 3; ++p) {
    float a00 = plane_axes[p * 9 + 0], a01 = plane_axes[p * 9 + 1], a02 = plane_axes[p * 9 + 2];
    float a10 = plane_axes[p * 9 + 3], a11 = plane_axes[p * 9 + 4], a12 = plane_axes[p * 9 + 5];
    float a20 = plane_axes[p * 9 + 6], a21 = plane_axes[p * 9 + 7], a22 = plane_axes[p * 9 + 8];
    float det = a00 * (a11 * a22 - a12 * a21) - a01 * (a10 * a22 - a12 * a20) + a02 * (a10 * a21 - a11 * a20);
    float id = 1.f / det;
    inv[p][0][0] =  (a11 * a22 - a12 * a21) * id;
    inv[p][0][1] = -(a01 * a22 - a02 * a21) * id;
    inv[p][1][0] = -(a10 * a22 - a12 * a20) * id;
    inv[p][1][1] =  (a00 * a22 - a02 * a20) * id;
    inv[p][2][0] =  (a10 * a21 - a11 * a20) * id;
    inv[p][2][1] = -(a00 * a21 - a01 * a20) * id;
  }

  const int ocol = lane & 15;
  const int rowoff = (lane >> 4) << 3;   // D-matrix: row = r + 8*(lane>=16)
  const int m = lane & 15;

  float Tcarry = 1.f;
  float fsum[16];
#pragma unroll
  for (int i = 0; i < 16; ++i) fsum[i] = 0.f;
  float rs0 = 0.f, rs1 = 0.f, rs2 = 0.f, xs = 0.f, ys = 0.f, zs = 0.f, msum = 0.f;

  v8f acc[16];

  // LDS byte offset of the weight slab (flat LDS address: low 32 bits = offset)
  const unsigned sW_lds = (unsigned)(unsigned long long)(void*)&sW[0];

  // --- per-layer GEMM: TDM weight staging + 16x16x32 f16 WMMA (pipelined) ---
  auto run_layer = [&](int Kt, const _Float16* Wlayer) {
#pragma unroll
    for (int nt = 0; nt < 16; ++nt) acc[nt] = (v8f){0.f, 0.f, 0.f, 0.f, 0.f, 0.f, 0.f, 0.f};
    for (int kt = 0; kt < Kt; ++kt) {
      __syncthreads();
      if (wave == 0) {
        // Tensor DMA: contiguous 16KB slab (4096 dwords) global -> LDS.
        unsigned long long ga = (unsigned long long)(const void*)(Wlayer + (size_t)kt * 8192);
        u32x4 g0;
        g0[0] = 1u;                                   // count=1 (valid user descriptor)
        g0[1] = sW_lds;                               // lds_addr (bytes)
        g0[2] = (unsigned)(ga & 0xffffffffull);       // global_addr[31:0]
        g0[3] = (unsigned)((ga >> 32) & 0x1ffffffull) // global_addr[56:32]
                | (2u << 30);                         // type=2 ("image")
        // data_size=4B; tensor_dim0=4096, tensor_dim1=1; tile_dim0=4096,
        // tile_dim1=1; tensor_dim0_stride=4096
        i32x8 g1 = {0x20000, 0x10000000, 0x10000, 0x10000000, 1, 4096, 0, 0};
        i32x4 g2 = {0, 0, 0, 0};
        i32x4 g3 = {0, 0, 0, 0};
#if __clang_major__ >= 23
        i32x8 g4 = {0, 0, 0, 0, 0, 0, 0, 0};
        __builtin_amdgcn_tensor_load_to_lds(g0, g1, g2, g3, g4, 0);
#else
        __builtin_amdgcn_tensor_load_to_lds(g0, g1, g2, g3, 0);
#endif
        if (kt + 1 < Kt)  // global_prefetch_b8 of next slab into GL2
          __builtin_prefetch(Wlayer + (size_t)(kt + 1) * 8192, 0, 1);
        __builtin_amdgcn_s_wait_tensorcnt(0);
      }
      __syncthreads();
      // A fragment (16x32 f16, contiguous K layout matching B): lane m=lane&15, +16 K-offset for upper half-wave
      v16h a;
      const _Float16* arow = act + m * 320 + kt * 32 + rowoff * 2;  // K offset (0/16)
#pragma unroll
      for (int v = 0; v < 8; ++v) {
        int kb = 2 * v;
        a[2 * v]     = arow[kb];
        a[2 * v + 1] = arow[kb + 1];
      }
      // software-pipelined B-fragment stream: load nt+1 before WMMA nt so the
      // ds_loads overlap XDL execution instead of s_wait_dscnt 0 per WMMA
      const _Float16* bptr = sW + (lane << 4);
      v16h bcur = *(const v16h*)bptr;
#pragma unroll
      for (int nt = 0; nt < 16; ++nt) {
        v16h bnxt = bcur;
        if (nt + 1 < 16) bnxt = *(const v16h*)(bptr + ((nt + 1) << 9));
        acc[nt] = __builtin_amdgcn_wmma_f32_16x16x32_f16(
            false, a, false, bcur, (short)0, acc[nt], false, false);
        bcur = bnxt;
      }
    }
  };

  // --- FiLM sin epilogue applied in accumulator layout; writeback to act rows ---
  auto film_epilogue = [&](const float* biasL, const float* gamL, const float* betL) {
#pragma unroll
    for (int nt = 0; nt < 16; ++nt) {
      int o = nt * 16 + ocol;
      float bo = biasL[o];
      float go = gamL[o];
      float be = betL[o];
#pragma unroll
      for (int r = 0; r < 8; ++r) {
        float hv = acc[nt][r] + bo;
        float xv = __builtin_sinf(go * hv + be);
        acc[nt][r] = xv;
        act[(r + rowoff) * 320 + o] = (_Float16)xv;
      }
    }
  };

  const float* gbase = gam + (size_t)bidx * 4 * 256;
  const float* bbase = bet + (size_t)bidx * 4 * 256;

  for (int tile = 0; tile < 2; ++tile) {
    // ================= Phase A: zero act tile, inputs, triplane sampling =================
    __syncthreads();
    {
      uint32_t* actw = (uint32_t*)act;
      for (int i = lane; i < 16 * 320 / 2; i += 32) actw[i] = 0u;
    }
    const int s = tile * 16 + m;
    const bool valid = (s < NSAMP);
    float px = 0.f, py = 0.f, pz = 0.f, nx = 0.f, ny = 0.f, nz = 0.f;
    if (valid) {
      int pt = ray * NSAMP + s;
      px = pts[pt * 3 + 0]; py = pts[pt * 3 + 1]; pz = pts[pt * 3 + 2];
      nx = (px - nearv) * inv_fn - 1.f;
      ny = (py - nearv) * inv_fn - 1.f;
      nz = (pz - nearv) * inv_fn - 1.f;
    }
    if (lane < 16 && valid) {
      act[m * 320 + 0] = (_Float16)nx;
      act[m * 320 + 1] = (_Float16)ny;
      act[m * 320 + 2] = (_Float16)nz;
      act[m * 320 + 288] = (_Float16)vdx;
      act[m * 320 + 289] = (_Float16)vdy;
      act[m * 320 + 290] = (_Float16)vdz;
    }
    if (valid) {
      // lane pair (l, l+16) splits the 32 emb channels for point m
      int ch0 = (lane >> 4) * 16;
      float emb[16];
#pragma unroll
      for (int i = 0; i < 16; ++i) emb[i] = 0.f;
#pragma unroll
      for (int p = 0; p < 3; ++p) {
        float X = nx * inv[p][0][0] + ny * inv[p][1][0] + nz * inv[p][2][0];
        float Y = nx * inv[p][0][1] + ny * inv[p][1][1] + nz * inv[p][2][1];
        float gx = ((X + 1.f) * (float)SSZ - 1.f) * 0.5f;
        float gy = ((Y + 1.f) * (float)SSZ - 1.f) * 0.5f;
        float x0f = __builtin_floorf(gx), y0f = __builtin_floorf(gy);
        float wx = gx - x0f, wy = gy - y0f;
        int x0 = (int)x0f, y0 = (int)y0f, x1 = x0 + 1, y1 = y0 + 1;
        bool vx0 = (x0 >= 0) && (x0 < SSZ), vx1 = (x1 >= 0) && (x1 < SSZ);
        bool vy0 = (y0 >= 0) && (y0 < SSZ), vy1 = (y1 >= 0) && (y1 < SSZ);
        int x0c = iminD(imaxD(x0, 0), SSZ - 1), x1c = iminD(imaxD(x1, 0), SSZ - 1);
        int y0c = iminD(imaxD(y0, 0), SSZ - 1), y1c = iminD(imaxD(y1, 0), SSZ - 1);
        float w00 = (1.f - wx) * (1.f - wy) * ((vx0 && vy0) ? 1.f : 0.f);
        float w01 = wx * (1.f - wy) * ((vx1 && vy0) ? 1.f : 0.f);
        float w10 = (1.f - wx) * wy * ((vx0 && vy1) ? 1.f : 0.f);
        float w11 = wx * wy * ((vx1 && vy1) ? 1.f : 0.f);
        const float* pb = planes + (size_t)(p * EMBC + ch0) * (SSZ * SSZ);
        int i00 = y0c * SSZ + x0c, i01 = y0c * SSZ + x1c;
        int i10 = y1c * SSZ + x0c, i11 = y1c * SSZ + x1c;
#pragma unroll
        for (int c = 0; c < 16; ++c) {
          const float* fb = pb + (size_t)c * (SSZ * SSZ);
          emb[c] += w00 * fb[i00] + w01 * fb[i01] + w10 * fb[i10] + w11 * fb[i11];
        }
      }
#pragma unroll
      for (int c = 0; c < 16; ++c)
        act[m * 320 + 256 + ch0 + c] = (_Float16)(emb[c] * (1.f / 3.f));
    }

    // ================= Phase B: FiLM-SIREN MLP via WMMA =================
    run_layer(1, Wp + WP_L0);
    film_epilogue(bias0, gbase + 0 * 256, bbase + 0 * 256);
    run_layer(8, Wp + WP_L1);
    film_epilogue(bias1, gbase + 1 * 256, bbase + 1 * 256);
    run_layer(8, Wp + WP_L2);
    film_epilogue(bias2, gbase + 2 * 256, bbase + 2 * 256);

    // sigma/sdf: dot over [x(256), emb(32)] = 288 (f16 LDS row)
    __syncthreads();
    float sdf = 0.f;
    if (lane < 16) {
      const _Float16* rowp = act + m * 320;
      float d = sigB[0];
      for (int k = 0; k < 288; ++k) d += (float)rowp[k] * sigW[k];
      sdf = d;
      if (valid) out[OUT_SDF + ray * NSAMP + s] = d;
    }

    // views layer: K padded 291->320 (x | emb | vdir | zeros)
    run_layer(10, Wp + WP_LV);
    film_epilogue(biasv, gbase + 3 * 256, bbase + 3 * 256);  // features now in acc + act rows

    // rgb head: 3 dots of 256 from the feature row
    float r0 = 0.f, r1 = 0.f, r2 = 0.f;
    if (lane < 16) {
      const _Float16* rowp = act + m * 320;
      float c0 = rgbB[0], c1 = rgbB[1], c2 = rgbB[2];
      for (int k = 0; k < 256; ++k) {
        float xv = (float)rowp[k];
        c0 += xv * rgbW[k];
        c1 += xv * rgbW[256 + k];
        c2 += xv * rgbW[512 + k];
      }
      r0 = c0; r1 = c1; r2 = c2;
    }

    // ================= Phase C: SDF volume integration for this tile =================
    float alpha = 0.f;
    if (lane < 16) {
      if (valid) {
        float sg = (1.f / (1.f + __builtin_expf(sdf / beta))) / beta;  // sigmoid(-sdf/b)/b
        float dz = (s < NSAMP - 1) ? (z_vals[ray * NSAMP + s + 1] - z_vals[ray * NSAMP + s]) : 1e10f;
        alpha = 1.f - __builtin_expf(-sg * dz * nrd);
      }
      sScan[wave][m] = 1.f - alpha + 1e-10f;
    }
    __syncthreads();
    float T = Tcarry, Tall = Tcarry;
    for (int j = 0; j < 16; ++j) {
      float f = sScan[wave][j];
      if (j < m) T *= f;
      Tall *= f;
    }
    float w = alpha * T;
    if (lane < 16) sWgt[wave][m] = w;
    Tcarry = Tall;
    __syncthreads();
    // weighted feature accumulation straight from the WMMA accumulator layout
#pragma unroll
    for (int nt = 0; nt < 16; ++nt) {
#pragma unroll
      for (int r = 0; r < 8; ++r)
        fsum[nt] += sWgt[wave][r + rowoff] * acc[nt][r];
    }
    if (lane < 16 && valid) {
      rs0 += w * r0; rs1 += w * r1; rs2 += w * r2;
      xs += w * px; ys += w * py; zs += w * pz;
      msum += w;
    }
  }  // tile loop

  // ================= Final: reductions + output writes =================
#pragma unroll
  for (int nt = 0; nt < 16; ++nt) {
    float v = fsum[nt] + __shfl_xor(fsum[nt], 16);
    if (lane < 16) out[OUT_FEAT + (size_t)ray * 256 + nt * 16 + lane] = v;
  }
#pragma unroll
  for (int d = 8; d; d >>= 1) {
    rs0 += __shfl_xor(rs0, d); rs1 += __shfl_xor(rs1, d); rs2 += __shfl_xor(rs2, d);
    xs += __shfl_xor(xs, d);  ys += __shfl_xor(ys, d);  zs += __shfl_xor(zs, d);
    msum += __shfl_xor(msum, d);
  }
  if (lane == 0) {
    out[OUT_RGB + ray * 3 + 0] = rs0;
    out[OUT_RGB + ray * 3 + 1] = rs1;
    out[OUT_RGB + ray * 3 + 2] = rs2;
    out[OUT_XYZ + ray * 3 + 0] = xs;
    out[OUT_XYZ + ray * 3 + 1] = ys;
    out[OUT_XYZ + ray * 3 + 2] = zs;
    out[OUT_MASK + ray] = msum;
  }
}

// ---------------------------------------------------------------------------
extern "C" void kernel_launch(void* const* d_in, const int* in_sizes, int n_in,
                              void* d_out, int out_size, void* d_ws, size_t ws_size,
                              hipStream_t stream) {
  const float* pts        = (const float*)d_in[0];
  const float* rays_d     = (const float*)d_in[1];
  const float* viewdirs   = (const float*)d_in[2];
  const float* z_vals     = (const float*)d_in[3];
  const float* nearp      = (const float*)d_in[4];
  const float* farp       = (const float*)d_in[5];
  const float* styles     = (const float*)d_in[6];
  const float* planes     = (const float*)d_in[7];
  const float* plane_axes = (const float*)d_in[8];
  const float* sbeta      = (const float*)d_in[9];
  // params: 3x pts_linears {W,b,gW,gb,bW,bb}, views {W,b,gW,gb,bW,bb}, rgb {W,b}, sigma {W,b}
  const float* L0W = (const float*)d_in[10]; const float* L0b = (const float*)d_in[11];
  const float* L0gW = (const float*)d_in[12]; const float* L0gb = (const float*)d_in[13];
  const float* L0bW = (const float*)d_in[14]; const float* L0bb = (const float*)d_in[15];
  const float* L1W = (const float*)d_in[16]; const float* L1b = (const float*)d_in[17];
  const float* L1gW = (const float*)d_in[18]; const float* L1gb = (const float*)d_in[19];
  const float* L1bW = (const float*)d_in[20]; const float* L1bb = (const float*)d_in[21];
  const float* L2W = (const float*)d_in[22]; const float* L2b = (const float*)d_in[23];
  const float* L2gW = (const float*)d_in[24]; const float* L2gb = (const float*)d_in[25];
  const float* L2bW = (const float*)d_in[26]; const float* L2bb = (const float*)d_in[27];
  const float* LVW = (const float*)d_in[28]; const float* LVb = (const float*)d_in[29];
  const float* LVgW = (const float*)d_in[30]; const float* LVgb = (const float*)d_in[31];
  const float* LVbW = (const float*)d_in[32]; const float* LVbb = (const float*)d_in[33];
  const float* rgbW = (const float*)d_in[34]; const float* rgbB = (const float*)d_in[35];
  const float* sigW = (const float*)d_in[36]; const float* sigB = (const float*)d_in[37];

  // workspace: [gamma 2048 f32][beta 2048 f32][Wp 221184 f16]
  float* gam = (float*)d_ws;
  float* bet = gam + 2048;
  _Float16* Wp = (_Float16*)(bet + 2048);
  float* out = (float*)d_out;

  film_prep<<<16, 128, 0, stream>>>(styles,
                                    L0gW, L0gb, L0bW, L0bb,
                                    L1gW, L1gb, L1bW, L1bb,
                                    L2gW, L2gb, L2bW, L2bb,
                                    LVgW, LVgb, LVbW, LVbb,
                                    gam, bet);
  pack_weights<<<(WP_TOTAL + 127) / 128, 128, 0, stream>>>(L0W, L1W, L2W, LVW, Wp);
  render_kernel<<<NRAYS / 4, 128, 0, stream>>>(
      pts, rays_d, viewdirs, z_vals, nearp, farp, planes, plane_axes, sbeta,
      L0b, L1b, L2b, LVb, rgbW, rgbB, sigW, sigB, gam, bet, Wp, out);
}
